// GLOT_55430847922213
// MI455X (gfx1250) — matrix operand order repaired and checked by reading, not compiled, not run on hardware
//
#include <hip/hip_runtime.h>
#include <hip/hip_bf16.h>
#include <math.h>

typedef __bf16 v16bf __attribute__((ext_vector_type(16)));
typedef __bf16 bf16x8 __attribute__((ext_vector_type(8)));
typedef float  v8f    __attribute__((ext_vector_type(8)));

#define TAU 0.3f
#define NEG_SLOPE 0.2f

__device__ __forceinline__ v8f wmma_bf16(v16bf a, v16bf b, v8f c) {
  return __builtin_amdgcn_wmma_f32_16x16x32_bf16(false, a, false, b, (short)0, c, false, false);
}
__device__ __forceinline__ float lrelu(float x) { return x > 0.f ? x : NEG_SLOPE * x; }

// ---------------------------------------------------------------------------
// Row L2-normalize; emit bf16 copies of hidden and hn. One wave per row.
__global__ void k_normalize(const float* __restrict__ h, __bf16* __restrict__ hb,
                            __bf16* __restrict__ hnb, int rows, int D) {
  int row = blockIdx.x * (blockDim.x >> 5) + (threadIdx.x >> 5);
  int lane = threadIdx.x & 31;
  if (row >= rows) return;
  const float* hr = h + (size_t)row * D;
  float ss = 0.f;
  for (int k = lane; k < D; k += 32) { float v = hr[k]; ss += v * v; }
  for (int off = 16; off; off >>= 1) ss += __shfl_xor(ss, off, 32);
  float inv = 1.0f / fmaxf(sqrtf(ss), 1e-8f);
  for (int k = lane; k < D; k += 32) {
    float v = hr[k];
    hb [(size_t)row * D + k] = (__bf16)v;
    hnb[(size_t)row * D + k] = (__bf16)(v * inv);
  }
}

// f32 (K x N) -> bf16 transposed (N x K)
__global__ void k_cast_transpose(const float* __restrict__ in, __bf16* __restrict__ out,
                                 int K, int N) {
  int i = blockIdx.x * blockDim.x + threadIdx.x;
  if (i < K * N) {
    int k = i / N, n = i % N;
    out[(size_t)n * K + k] = (__bf16)in[i];
  }
}

// ae = dot(w_edge, att_edge), 128 elements
__global__ void k_dot128(const float* __restrict__ x, const float* __restrict__ y,
                         float* __restrict__ out) {
  __shared__ float s[128];
  int t = threadIdx.x;
  s[t] = x[t] * y[t];
  __syncthreads();
  for (int off = 64; off; off >>= 1) { if (t < off) s[t] += s[t + off]; __syncthreads(); }
  if (t == 0) *out = s[0];
}

// ---------------------------------------------------------------------------
// sim = hn @ hn^T per batch via bf16 WMMA; threshold+mask -> 1 bit/edge.
// Each wave: one 16x32 block (two 16x16 tiles sharing A) -> one u32 word/row.
__global__ void k_sim_adj(const __bf16* __restrict__ hn, const int* __restrict__ mask,
                          unsigned* __restrict__ adj, int L, int D) {
  int b = blockIdx.z;
  int wave = threadIdx.x >> 5, lane = threadIdx.x & 31;
  int i0 = blockIdx.x * 16;
  int j0 = blockIdx.y * 128 + wave * 32;
  const __bf16* hb = hn + (size_t)b * L * D;
  size_t bL = (size_t)b * L;
  int lh = lane >> 4;
  int rowA = i0 + (lane & 15);
  int kbA = lh ? 16 : 0;
  int colj = j0 + (lane & 15);
  int kbB = lh ? 16 : 0;
  v8f c0 = {}, c1 = {};
  for (int k0 = 0; k0 < D; k0 += 32) {
    union { v16bf v; bf16x8 h[2]; } a;
    const __bf16* pa = hb + (size_t)rowA * D + k0;
    a.h[0] = *(const bf16x8*)(pa + kbA);
    a.h[1] = *(const bf16x8*)(pa + 8 + kbA);
    v16bf b0 = *(const v16bf*)(hb + (size_t)colj * D + k0 + kbB);
    v16bf b1 = *(const v16bf*)(hb + (size_t)(colj + 16) * D + k0 + kbB);
    c0 = wmma_bf16(a.v, b0, c0);
    c1 = wmma_bf16(a.v, b1, c1);
  }
  bool vj0 = mask[bL + colj] != 0;
  bool vj1 = mask[bL + colj + 16] != 0;
  unsigned word = 0;
  #pragma unroll
  for (int r = 0; r < 8; r++) {
    int rr = i0 + r + (lh ? 8 : 0);
    bool vr = mask[bL + rr] != 0;
    unsigned m0 = (unsigned)__ballot((c0[r] > TAU) && vr && vj0);
    unsigned m1 = (unsigned)__ballot((c1[r] > TAU) && vr && vj1);
    if (lane == r)     word = (m0 & 0xffffu) | ((m1 & 0xffffu) << 16);
    if (lane == r + 8) word = (m0 >> 16) | (m1 & 0xffff0000u);
  }
  if (lane < 16)
    adj[(bL + i0 + lane) * (L >> 5) + (j0 >> 5)] = word;
}

// ---------------------------------------------------------------------------
// Projection GEMM, N fixed at 128: one wave owns a 16x128 row strip with
// 8 accumulators so the A tile is fetched once per K-chunk (8 WMMA / chunk).
__global__ void k_gemm_row128(const __bf16* __restrict__ A, const __bf16* __restrict__ WT,
                              __bf16* __restrict__ Cb, float* __restrict__ C, int K) {
  int lane = threadIdx.x & 31, wave = threadIdx.x >> 5;
  int m0 = (blockIdx.x * (blockDim.x >> 5) + wave) * 16;
  int lh = lane >> 4;
  int rowA = m0 + (lane & 15); int kbA = lh ? 16 : 0;
  int colC = lane & 15;        int kbB = lh ? 16 : 0;
  v8f acc[8];
  v8f zero = {};
  #pragma unroll
  for (int t = 0; t < 8; t++) acc[t] = zero;
  for (int k0 = 0; k0 < K; k0 += 32) {
    union { v16bf v; bf16x8 h[2]; } a;
    const __bf16* pa = A + (size_t)rowA * K + k0;
    a.h[0] = *(const bf16x8*)(pa + kbA);
    a.h[1] = *(const bf16x8*)(pa + 8 + kbA);
    #pragma unroll
    for (int t = 0; t < 8; t++) {
      v16bf bt = *(const v16bf*)(WT + (size_t)(t * 16 + colC) * K + k0 + kbB);
      acc[t] = wmma_bf16(a.v, bt, acc[t]);
    }
  }
  #pragma unroll
  for (int t = 0; t < 8; t++) {
    #pragma unroll
    for (int r = 0; r < 8; r++) {
      size_t idx = (size_t)(m0 + r + (lh ? 8 : 0)) * 128 + t * 16 + colC;
      if (Cb) Cb[idx] = (__bf16)acc[t][r];
      if (C)  C[idx]  = acc[t][r];
    }
  }
}

// al_s = g . a_src, al_d = g . a_dst (H=128). One wave per node.
__global__ void k_node_scores(const __bf16* __restrict__ g, const float* __restrict__ asrc,
                              const float* __restrict__ adst, float* __restrict__ als,
                              float* __restrict__ ald, int rows) {
  int row = blockIdx.x * (blockDim.x >> 5) + (threadIdx.x >> 5);
  int lane = threadIdx.x & 31;
  if (row >= rows) return;
  const __bf16* gr = g + (size_t)row * 128;
  float ss = 0.f, sd = 0.f;
  for (int k = lane; k < 128; k += 32) {
    float v = (float)gr[k];
    ss += v * asrc[k]; sd += v * adst[k];
  }
  for (int off = 16; off; off >>= 1) { ss += __shfl_xor(ss, off, 32); sd += __shfl_xor(sd, off, 32); }
  if (lane == 0) { als[row] = ss; ald[row] = sd; }
}

// ---------------------------------------------------------------------------
// Fused GAT attention-aggregate: masked softmax over incoming edges + attn@g.
// Block = 8 waves = 128 dst rows; g chunk staged transposed in LDS per 32-j step.
__global__ void k_gat_agg(const __bf16* __restrict__ g, const unsigned* __restrict__ adj,
                          const float* __restrict__ als, const float* __restrict__ ald,
                          const float* __restrict__ aep, const float* __restrict__ bias,
                          float* __restrict__ H, __bf16* __restrict__ Hb, int L) {
  __shared__ float s_als[1024];
  __shared__ __attribute__((aligned(32))) __bf16 s_gT[128 * 32];
  int b = blockIdx.y;
  int tid = threadIdx.x, wave = tid >> 5, lane = tid & 31;
  int i0 = blockIdx.x * 128 + wave * 16;
  size_t bL = (size_t)b * L;
  float ae = *aep;
  for (int l = tid; l < L; l += 256) s_als[l] = als[bL + l];
  __syncthreads();

  int lh = lane >> 4;
  int row = lane & 15;
  int grow = i0 + row;
  const unsigned* adjrow = adj + (bL + grow) * (L >> 5);
  float ald_r = ald[bL + grow];

  // pass 1: masked max of al_s (leakyrelu is monotone -> row max of logits)
  float m = -3.0e38f;
  for (int ch = lh; ch < (L >> 5); ch += 2) {
    unsigned w = adjrow[ch];
    while (w) {
      int t = __ffs(w) - 1;
      w &= w - 1;
      m = fmaxf(m, s_als[ch * 32 + t]);
    }
  }
  m = fmaxf(m, __shfl_xor(m, 16, 32));
  float M = lrelu(ald_r + m + ae);

  v8f acc[8];
  v8f zero = {};
  #pragma unroll
  for (int t = 0; t < 8; t++) acc[t] = zero;
  float ssum = 0.f;
  int kbA = lh ? 8 : 0;
  int colC = lane & 15;
  int kbB = lh ? 8 : 0;

  for (int jc = 0; jc < (L >> 5); jc++) {
    int j0 = jc * 32;
    __syncthreads();
    for (int f = tid; f < 32 * 128; f += 256) {
      int cc = f & 127, jj = f >> 7;
      s_gT[cc * 32 + jj] = g[(bL + j0 + jj) * 128 + cc];
    }
    __syncthreads();
    unsigned w = adjrow[jc];
    unsigned w0 = w & 0xffffu, w1 = w >> 16;
    v16bf a;
    #pragma unroll
    for (int e = 0; e < 8; e++) {
      int jj = kbA + e;
      float p = 0.f;
      if ((w0 >> jj) & 1u)
        p = __expf(lrelu(ald_r + s_als[j0 + jj] + ae) - M);
      a[e] = (__bf16)p; ssum += p;
      float p2 = 0.f;
      if ((w1 >> jj) & 1u)
        p2 = __expf(lrelu(ald_r + s_als[j0 + 16 + jj] + ae) - M);
      a[8 + e] = (__bf16)p2; ssum += p2;
    }
    #pragma unroll
    for (int t = 0; t < 8; t++) {
      union { v16bf v; bf16x8 h[2]; } bt;
      bt.h[0] = *(const bf16x8*)&s_gT[(16 * t + colC) * 32 + kbB];
      bt.h[1] = *(const bf16x8*)&s_gT[(16 * t + colC) * 32 + 16 + kbB];
      acc[t] = wmma_bf16(a, bt.v, acc[t]);
    }
  }
  ssum += __shfl_xor(ssum, 16, 32);
  #pragma unroll
  for (int r = 0; r < 8; r++) {
    int rr = r + lh * 8;
    float denom = fmaxf(__shfl(ssum, rr, 32), 1e-16f);
    #pragma unroll
    for (int t = 0; t < 8; t++) {
      int col = 16 * t + colC;
      float o = fmaxf(acc[t][r] / denom + bias[col], 0.f);
      size_t idx = (bL + i0 + rr) * 128 + col;
      H[idx] = o;
      Hb[idx] = (__bf16)o;
    }
  }
}

// ---------------------------------------------------------------------------
// scores = tanh(h_all @ S1_W + S1_b) @ S2_W + S2_b ; h_all = [hid|h1|h2] (K=1024).
// 4 column tiles per wave share each A fetch (4 accumulators).
__global__ void k_score(const __bf16* __restrict__ hidb, const __bf16* __restrict__ h1b,
                        const __bf16* __restrict__ h2b, const __bf16* __restrict__ S1WT,
                        const float* __restrict__ S1b, const float* __restrict__ S2W,
                        const float* __restrict__ S2b, float* __restrict__ scores) {
  int lane = threadIdx.x & 31, wave = threadIdx.x >> 5;
  int m0 = (blockIdx.x * (blockDim.x >> 5) + wave) * 16;
  int lh = lane >> 4;
  int rowA = m0 + (lane & 15);
  int kbA = lh ? 16 : 0, kbB = lh ? 16 : 0;
  int colC = lane & 15;
  v8f sacc = {};
  v8f zero = {};
  for (int cg = 0; cg < 8; cg++) {
    v8f c[4];
    #pragma unroll
    for (int u = 0; u < 4; u++) c[u] = zero;
    for (int kc = 0; kc < 32; kc++) {
      int k0 = kc * 32;
      const __bf16* pa; int ko;
      if (k0 < 768)      { pa = hidb + (size_t)rowA * 768; ko = k0; }
      else if (k0 < 896) { pa = h1b  + (size_t)rowA * 128; ko = k0 - 768; }
      else               { pa = h2b  + (size_t)rowA * 128; ko = k0 - 896; }
      union { v16bf v; bf16x8 h[2]; } a;
      a.h[0] = *(const bf16x8*)(pa + ko + kbA);
      a.h[1] = *(const bf16x8*)(pa + ko + 8 + kbA);
      #pragma unroll
      for (int u = 0; u < 4; u++) {
        int col = cg * 64 + u * 16 + colC;
        v16bf bt = *(const v16bf*)(S1WT + (size_t)col * 1024 + k0 + kbB);
        c[u] = wmma_bf16(a.v, bt, c[u]);
      }
    }
    #pragma unroll
    for (int u = 0; u < 4; u++) {
      int col = cg * 64 + u * 16 + colC;
      float s2 = S2W[col], b1 = S1b[col];
      #pragma unroll
      for (int r = 0; r < 8; r++) sacc[r] += tanhf(c[u][r] + b1) * s2;
    }
  }
  #pragma unroll
  for (int off = 1; off < 16; off <<= 1) {
    #pragma unroll
    for (int r = 0; r < 8; r++) { float t = __shfl_xor(sacc[r], off, 32); sacc[r] += t; }
  }
  float s2b = *S2b;
  if (lane == 0)  { for (int r = 0; r < 8; r++) scores[m0 + r]     = sacc[r] + s2b; }
  if (lane == 16) { for (int r = 0; r < 8; r++) scores[m0 + 8 + r] = sacc[r] + s2b; }
}

// masked softmax over L per batch + weighted pooling of h_all. One block/batch.
__global__ void k_pool(const float* __restrict__ scores, const int* __restrict__ mask,
                       const float* __restrict__ hid, const float* __restrict__ h1,
                       const float* __restrict__ h2, float* __restrict__ out, int L) {
  __shared__ float sw[1024];
  __shared__ float red[1024];
  int b = blockIdx.x, t = threadIdx.x;
  size_t bL = (size_t)b * L;
  bool v = mask[bL + t] != 0;
  float sc = v ? scores[bL + t] : -3.0e38f;
  red[t] = sc; __syncthreads();
  for (int off = 512; off; off >>= 1) { if (t < off) red[t] = fmaxf(red[t], red[t + off]); __syncthreads(); }
  float mx = red[0]; __syncthreads();
  float e = v ? __expf(sc - mx) : 0.f;
  red[t] = e; __syncthreads();
  for (int off = 512; off; off >>= 1) { if (t < off) red[t] += red[t + off]; __syncthreads(); }
  sw[t] = e / fmaxf(red[0], 1e-16f);
  __syncthreads();
  float acc = 0.f;
  if (t < 768) {
    const float* p = hid + bL * 768 + t;
    for (int l = 0; l < L; l++) acc += sw[l] * p[(size_t)l * 768];
  } else if (t < 896) {
    const float* p = h1 + bL * 128 + (t - 768);
    for (int l = 0; l < L; l++) acc += sw[l] * p[(size_t)l * 128];
  } else {
    const float* p = h2 + bL * 128 + (t - 896);
    for (int l = 0; l < L; l++) acc += sw[l] * p[(size_t)l * 128];
  }
  out[(size_t)b * 1024 + t] = acc;
}

// ---------------------------------------------------------------------------
extern "C" void kernel_launch(void* const* d_in, const int* in_sizes, int n_in,
                              void* d_out, int out_size, void* d_ws, size_t ws_size,
                              hipStream_t stream) {
  const float* hidden = (const float*)d_in[0];
  const int*   amask  = (const int*)  d_in[1];
  const float* W0     = (const float*)d_in[2];
  const float* asrc0  = (const float*)d_in[3];
  const float* adst0  = (const float*)d_in[4];
  const float* wedge0 = (const float*)d_in[5];
  const float* aedge0 = (const float*)d_in[6];
  const float* bias0  = (const float*)d_in[7];
  const float* W1     = (const float*)d_in[8];
  const float* asrc1  = (const float*)d_in[9];
  const float* adst1  = (const float*)d_in[10];
  const float* wedge1 = (const float*)d_in[11];
  const float* aedge1 = (const float*)d_in[12];
  const float* bias1  = (const float*)d_in[13];
  const float* S1W    = (const float*)d_in[14];
  const float* S1b    = (const float*)d_in[15];
  const float* S2W    = (const float*)d_in[16];
  const float* S2b    = (const float*)d_in[17];
  float* out = (float*)d_out;

  const int B = 16, L = 1024, D = 768, Hh = 128;
  const int rows = B * L;

  size_t off = 0;
  auto alloc = [&](size_t bytes) -> char* {
    char* p = (char*)d_ws + off;
    off += (bytes + 255) & ~(size_t)255;
    return p;
  };
  __bf16*   hidb = (__bf16*)alloc((size_t)rows * D * 2);
  __bf16*   hnb  = (__bf16*)alloc((size_t)rows * D * 2);
  unsigned* adj  = (unsigned*)alloc((size_t)rows * (L / 32) * 4);
  __bf16*   W0T  = (__bf16*)alloc((size_t)Hh * D * 2);
  __bf16*   W1T  = (__bf16*)alloc((size_t)Hh * Hh * 2);
  __bf16*   S1WT = (__bf16*)alloc((size_t)512 * 1024 * 2);
  __bf16*   g0b  = (__bf16*)alloc((size_t)rows * Hh * 2);
  __bf16*   g1b  = (__bf16*)alloc((size_t)rows * Hh * 2);
  float*    h1   = (float*) alloc((size_t)rows * Hh * 4);
  __bf16*   h1b  = (__bf16*)alloc((size_t)rows * Hh * 2);
  float*    h2   = (float*) alloc((size_t)rows * Hh * 4);
  __bf16*   h2b  = (__bf16*)alloc((size_t)rows * Hh * 2);
  float*    als0 = (float*) alloc((size_t)rows * 4);
  float*    ald0 = (float*) alloc((size_t)rows * 4);
  float*    als1 = (float*) alloc((size_t)rows * 4);
  float*    ald1 = (float*) alloc((size_t)rows * 4);
  float*    scrs = (float*) alloc((size_t)rows * 4);
  float*    ae0  = (float*) alloc(256);
  float*    ae1  = (float*) alloc(256);

  // precompute: bf16 casts / transposes / edge scalar
  k_normalize<<<dim3(rows / 8), dim3(256), 0, stream>>>(hidden, hidb, hnb, rows, D);
  k_cast_transpose<<<dim3((D * Hh + 255) / 256), dim3(256), 0, stream>>>(W0, W0T, D, Hh);
  k_cast_transpose<<<dim3((Hh * Hh + 255) / 256), dim3(256), 0, stream>>>(W1, W1T, Hh, Hh);
  k_cast_transpose<<<dim3((1024 * 512 + 255) / 256), dim3(256), 0, stream>>>(S1W, S1WT, 1024, 512);
  k_dot128<<<dim3(1), dim3(128), 0, stream>>>(wedge0, aedge0, ae0);
  k_dot128<<<dim3(1), dim3(128), 0, stream>>>(wedge1, aedge1, ae1);

  // cosine-similarity graph -> 1-bit adjacency
  k_sim_adj<<<dim3(L / 16, L / 128, B), dim3(128), 0, stream>>>(hnb, amask, adj, L, D);

  // GAT layer 0
  k_gemm_row128<<<dim3(rows / 64), dim3(128), 0, stream>>>(hidb, W0T, g0b, (float*)nullptr, D);
  k_node_scores<<<dim3(rows / 8), dim3(256), 0, stream>>>(g0b, asrc0, adst0, als0, ald0, rows);
  k_gat_agg<<<dim3(L / 128, B), dim3(256), 0, stream>>>(g0b, adj, als0, ald0, ae0, bias0, h1, h1b, L);

  // GAT layer 1
  k_gemm_row128<<<dim3(rows / 64), dim3(128), 0, stream>>>(h1b, W1T, g1b, (float*)nullptr, Hh);
  k_node_scores<<<dim3(rows / 8), dim3(256), 0, stream>>>(g1b, asrc1, adst1, als1, ald1, rows);
  k_gat_agg<<<dim3(L / 128, B), dim3(256), 0, stream>>>(g1b, adj, als1, ald1, ae1, bias1, h2, h2b, L);

  // scoring MLP + masked-softmax pooling
  k_score<<<dim3(rows / 64), dim3(128), 0, stream>>>(hidb, h1b, h2b, S1WT, S1b, S2W, S2b, scrs);
  k_pool<<<dim3(B), dim3(1024), 0, stream>>>(scrs, amask, hidden, h1, h2, out, L);
}
